// LTCCell_88768384074316
// MI455X (gfx1250) — matrix operand order repaired and checked by reading, not compiled, not run on hardware
//
#include <hip/hip_runtime.h>
#include <hip/hip_bf16.h>
#include <hip/hip_fp16.h>

// LTC cell for MI455X (gfx1250): bf16 WMMA GEMMs with f32 accumulation.
//   out = hs [B=32, S=2048, H=512], f32.
// Phase 0: Wh,Wx f32 -> bf16 into d_ws (768 KB).
// Phase 1: xproj = x @ Wx^T via v_wmma_f32_16x16x32_bf16, written into d_out.
// Phase 2: persistent 1-block scan. Dynamic LDS (224 KB < 320 KB/WGP):
//   - h state f32[32][512]           (64 KB)
//   - swizzled bf16 tanh(A) buffer   (32 KB)
//   - double-buffered xproj slab     (2 x 64 KB), filled one step ahead by
//     the Tensor Data Mover (tensor_load_to_lds, TENSORcnt), so the 2048-step
//     critical path loads only Wh (bf16 B operands) from global, plus h_t stores.

#define DT 1.0f

typedef __bf16 v16bf __attribute__((ext_vector_type(16)));
typedef float  v8f   __attribute__((ext_vector_type(8)));
typedef unsigned int u32x4 __attribute__((ext_vector_type(4)));
typedef int          i32x8 __attribute__((ext_vector_type(8)));
typedef int          i32x4 __attribute__((ext_vector_type(4)));

// ---------------------------------------------------------------------------
// Phase 0: weight conversion f32 -> bf16 into workspace.
// ws layout (bf16 elems): [0, 512*512) = Wh ; [512*512, 512*512+512*256) = Wx
// ---------------------------------------------------------------------------
__global__ void cvt_weights_bf16(const float* __restrict__ Wh,
                                 const float* __restrict__ Wx,
                                 __bf16* __restrict__ wsb) {
    int i = blockIdx.x * 256 + threadIdx.x;          // grid = 1024 blocks
    if (i < 512 * 512) wsb[i] = (__bf16)Wh[i];
    if (i < 512 * 256) wsb[512 * 512 + i] = (__bf16)Wx[i];
}

// ---------------------------------------------------------------------------
// Phase 1: xproj[r, n] = sum_k x[r, k] * Wx[n, k]   (r = b*S+s, 65536 rows)
// One block = one 16-row m-tile; 8 waves x 4 n-tiles = full N=512.
// ---------------------------------------------------------------------------
__global__ __launch_bounds__(256) void xproj_wmma(
    const float* __restrict__ x,          // [65536, 256]
    const __bf16* __restrict__ Wxb,       // [512, 256] bf16
    float* __restrict__ out)              // [65536, 512] (xproj)
{
    const int lane = threadIdx.x & 31;
    const int w    = threadIdx.x >> 5;    // wave 0..7
    const int half = lane >> 4;           // 0|1
    const int n    = lane & 15;           // col-in-tile (B/C/D) / row (A)
    const size_t mt = blockIdx.x;         // m-tile 0..4095

    const float* arow = x + (mt * 16 + (size_t)n) * 256;

    v8f c[4];
    #pragma unroll
    for (int j = 0; j < 4; ++j) c[j] = (v8f){};

    for (int kk = 0; kk < 8; ++kk) {      // K = 256 = 8 * 32
        // A chunk: contiguous K layout: element 2i+e -> K=half*16+2i+e
        //          (matches B)
        v16bf a;
        const float* ak = arow + kk * 32;
        #pragma unroll
        for (int i = 0; i < 8; ++i) {
            const int kbase = half * 16 + 2 * i;
            const float2 f2 = *(const float2*)(ak + kbase);
            a[2 * i]     = (__bf16)f2.x;
            a[2 * i + 1] = (__bf16)f2.y;
        }
        #pragma unroll
        for (int j = 0; j < 4; ++j) {
            const int nt = w * 4 + j;
            // B lane (half,n): K = half*16 + slot -> contiguous row of Wx
            const v16bf b = *(const v16bf*)(Wxb + (size_t)(nt * 16 + n) * 256 +
                                            kk * 32 + half * 16);
            c[j] = __builtin_amdgcn_wmma_f32_16x16x32_bf16(
                false, a, false, b, (short)0, c[j], false, false);
        }
    }

    #pragma unroll
    for (int j = 0; j < 4; ++j) {
        const int nt = w * 4 + j;
        #pragma unroll
        for (int r = 0; r < 8; ++r) {
            const size_t row = mt * 16 + r + 8 * half;
            out[row * 512 + nt * 16 + n] = c[j][r];
        }
    }
}

// ---------------------------------------------------------------------------
// TDM helper: issue tensor_load_to_lds of the 2D xproj slab for timestep t:
//   tile = out[b][t][col], b=0..31, col=0..511  (row stride 2048*512 f32)
// into LDS byte offset lds_off. Wave-uniform arguments only.
// D# packing per CDNA5 ISA ch.8 (groups 0/1; groups 2/3 zero: 2D tile).
// This toolchain's builtin is the 6-arg form:
//   (u32x4 g0, i32x8 g1, i32x4, i32x4, i32x8, i32 cpol)
// ---------------------------------------------------------------------------
__device__ __forceinline__ void tdm_load_xslab(const float* out, int t,
                                               unsigned int lds_off) {
    const unsigned long long gaddr =
        (unsigned long long)(const void*)(out + (size_t)t * 512);

    u32x4 g0;
    g0.x = 1u;                               // count=1, user descriptor
    g0.y = lds_off;                          // lds_addr (bytes)
    g0.z = (unsigned int)gaddr;              // global_addr[31:0]
    g0.w = (unsigned int)((gaddr >> 32) & 0x01FFFFFFull) | (2u << 30); // type=2

    i32x8 g1;
    g1[0] = (int)(2u << 16);                 // data_size=2 (4 bytes)
    g1[1] = (int)(512u << 16);               // tensor_dim0 = 512 (low 16)
    g1[2] = (int)(32u << 16);                // tensor_dim1 = 32  (low 16)
    g1[3] = (int)(512u << 16);               // tile_dim0 = 512
    g1[4] = 32;                              // tile_dim1 = 32, tile_dim2 = 0
    g1[5] = (int)(2048u * 512u);             // tensor_dim0_stride[31:0]
    g1[6] = 0;                               // stride[47:32]=0, dim1_stride lo=0
    g1[7] = 0;

    const i32x4 gz4 = (i32x4){0, 0, 0, 0};   // groups 2/3 unused (2D tile)
    const i32x8 gz8 = (i32x8){0, 0, 0, 0, 0, 0, 0, 0};
    __builtin_amdgcn_tensor_load_to_lds(g0, g1, gz4, gz4, gz8, 0);
}

// ---------------------------------------------------------------------------
// Phase 2: sequential scan, single workgroup (1024 thr = 32 waves).
//   h_{t} = h_{t-1} + DT*(-h_{t-1} + xp_t + tanh(h_{t-1})@Wh^T + bias)/tau
// Tile ownership: wave w -> m-tile (w&1), n-tiles {2*(w>>1), 2*(w>>1)+1}.
// ---------------------------------------------------------------------------
__global__ __launch_bounds__(1024) void ltc_scan(
    const __bf16* __restrict__ Whb,       // [512, 512] bf16
    const float* __restrict__ tau,        // [512]
    const float* __restrict__ bias,       // [512]
    float* __restrict__ out)              // [32, 2048, 512]; in: xproj, out: h
{
    extern __shared__ char smem[];
    float*  hbuf = (float*)smem;                       // 64 KB h state
    __bf16* abuf = (__bf16*)(smem + 64 * 1024);        // 32 KB swizzled A
    float*  xbuf0 = (float*)(smem + 96 * 1024);        // 64 KB xp slab buf 0
    float*  xbuf1 = (float*)(smem + 160 * 1024);       // 64 KB xp slab buf 1
    const unsigned int xoff0 =
        (unsigned int)(unsigned long long)(void*)xbuf0; // LDS byte offsets
    const unsigned int xoff1 =
        (unsigned int)(unsigned long long)(void*)xbuf1;

    const int tid  = threadIdx.x;
    const int lane = tid & 31;
    const int w    = tid >> 5;            // 0..31
    const int half = lane >> 4;
    const int n    = lane & 15;
    const int mt   = w & 1;
    const int q    = w >> 1;              // 0..15
    const int nt0  = 2 * q, nt1 = 2 * q + 1;

    const float it0 = 1.0f / tau[nt0 * 16 + n];
    const float it1 = 1.0f / tau[nt1 * 16 + n];
    const float bs0 = bias[nt0 * 16 + n];
    const float bs1 = bias[nt1 * 16 + n];

    // zero-init h state (h0 = 0); step 0 then needs no special casing
    for (int i = tid; i < 32 * 512; i += 1024) hbuf[i] = 0.0f;

    // TDM-prefetch xp slab for t=0 into buffer 0 (wave 0 issues + drains)
    if (w == 0) {
        tdm_load_xslab(out, 0, xoff0);
        __builtin_amdgcn_s_wait_tensorcnt(0);
    }
    __syncthreads();

    // tanh-pass mapping: this thread owns h[trow][tk0..tk0+15]
    const int trow = tid >> 5;            // 0..31 (batch)
    const int tk0  = (tid & 31) * 16;     // 0..496

    for (int t = 0; t < 2048; ++t) {
        const float* xcur = (t & 1) ? xbuf1 : xbuf0;

        // ---- tanh pass: h (LDS) -> swizzled bf16 A-buffer (LDS)
        {
            const int mt_  = trow >> 4;
            const int mloc = trow & 15;
            #pragma unroll
            for (int v4 = 0; v4 < 4; ++v4) {
                const float4 hv = *(const float4*)&hbuf[trow * 512 + tk0 + 4 * v4];
                #pragma unroll
                for (int e4 = 0; e4 < 4; ++e4) {
                    const int k = tk0 + 4 * v4 + e4;
                    const float tv =
                        tanhf(e4 == 0 ? hv.x : e4 == 1 ? hv.y : e4 == 2 ? hv.z
                                                                        : hv.w);
                    // invert A layout: k%32 -> (half, vgpr, elem)
                    const int kp  = k & 31;
                    const int hi  = (kp >= 16) ? 1 : 0;
                    const int kq  = kp - (hi << 4);
                    const int hf  = kq >> 3;
                    const int rem = kq & 7;
                    const int ii  = (hi << 2) + (rem >> 1);
                    const int e   = rem & 1;
                    const int ln  = (hi << 4) + mloc;
                    const int slot = kq;
                    abuf[(size_t)(((mt_ * 16 + (k >> 5)) * 32) + ln) * 16 +
                         slot] = (__bf16)tv;
                }
            }
        }
        // kick off TDM for next step's xp slab; overlaps with GEMM below
        if (w == 0 && t + 1 < 2048)
            tdm_load_xslab(out, t + 1, (t & 1) ? xoff0 : xoff1);
        __syncthreads();

        // ---- GEMM: c[m-tile, n-tile] += A(16x32) x B(32x16) over K=512
        v8f c0 = (v8f){}, c1 = (v8f){};
        #pragma unroll
        for (int kk = 0; kk < 16; ++kk) {
            const v16bf a =
                *(const v16bf*)&abuf[(size_t)(((mt * 16 + kk) * 32) + lane) * 16];
            const v16bf b0 = *(const v16bf*)(Whb + (size_t)(nt0 * 16 + n) * 512 +
                                             kk * 32 + half * 16);
            const v16bf b1 = *(const v16bf*)(Whb + (size_t)(nt1 * 16 + n) * 512 +
                                             kk * 32 + half * 16);
            c0 = __builtin_amdgcn_wmma_f32_16x16x32_bf16(
                false, a, false, b0, (short)0, c0, false, false);
            c1 = __builtin_amdgcn_wmma_f32_16x16x32_bf16(
                false, a, false, b1, (short)0, c1, false, false);
        }

        // ---- Euler update epilogue; lane col fixed, rows r + 8*half.
        //      hp and xp from LDS; only the h_t store touches global.
        #pragma unroll
        for (int r = 0; r < 8; ++r) {
            const int b = mt * 16 + r + 8 * half;   // batch row
            {
                const int col = nt0 * 16 + n;
                const float hp = hbuf[b * 512 + col];
                const float xp = xcur[b * 512 + col];
                const float hn = hp + DT * ((xp + c0[r] + bs0 - hp) * it0);
                hbuf[b * 512 + col] = hn;
                out[((size_t)b * 2048 + t) * 512 + col] = hn;
            }
            {
                const int col = nt1 * 16 + n;
                const float hp = hbuf[b * 512 + col];
                const float xp = xcur[b * 512 + col];
                const float hn = hp + DT * ((xp + c1[r] + bs1 - hp) * it1);
                hbuf[b * 512 + col] = hn;
                out[((size_t)b * 2048 + t) * 512 + col] = hn;
            }
        }
        // drain the next-step TDM before the barrier: after the barrier all
        // waves may consume the freshly filled slab
        if (w == 0) __builtin_amdgcn_s_wait_tensorcnt(0);
        __syncthreads();
    }
}

// ---------------------------------------------------------------------------
extern "C" void kernel_launch(void* const* d_in, const int* in_sizes, int n_in,
                              void* d_out, int out_size, void* d_ws,
                              size_t ws_size, hipStream_t stream) {
    const float* x    = (const float*)d_in[0];   // [32,2048,256]
    const float* Wx   = (const float*)d_in[1];   // [512,256]
    const float* Wh   = (const float*)d_in[2];   // [512,512]
    const float* tau  = (const float*)d_in[3];   // [512]
    const float* bias = (const float*)d_in[4];   // [512]
    float* out = (float*)d_out;                  // [32,2048,512]

    __bf16* wsb = (__bf16*)d_ws;                 // needs 768 KB
    __bf16* Whb = wsb;
    __bf16* Wxb = wsb + 512 * 512;

    cvt_weights_bf16<<<1024, 256, 0, stream>>>(Wh, Wx, wsb);
    xproj_wmma<<<4096, 256, 0, stream>>>(x, Wxb, out);
    // 224 KB dynamic LDS: 64 KB h + 32 KB A buffer + 2 x 64 KB TDM xp slabs
    ltc_scan<<<1, 1024, 224 * 1024, stream>>>(Whb, tau, bias, out);
}
